// MultiHeadAttention_76905684402660
// MI455X (gfx1250) — matrix operand order repaired and checked by reading, not compiled, not run on hardware
//
#include <hip/hip_runtime.h>

typedef __attribute__((ext_vector_type(16))) __bf16 v16bf;
typedef __attribute__((ext_vector_type(8)))  __bf16 v8bf;
typedef __attribute__((ext_vector_type(8)))  float  v8f;

static constexpr int Bb = 2, Tt = 2048, Hh = 512, NH = 16, HD = 32;
static constexpr size_t PER_MAT = (size_t)Bb * NH * Tt * HD;  // bf16 elements per q/k/v

__device__ __forceinline__ v8f wmma_bf16(v16bf a, v16bf b, v8f c) {
  // D = A(16x32 bf16) * B(32x16 bf16) + C(16x16 f32)
  return __builtin_amdgcn_wmma_f32_16x16x32_bf16(false, a, false, b, (short)0, c,
                                                 false, false);
}

// A-fragment (16x32, 16-bit): lane covers row M=lane&15.
// Layout: K = e + 16*hi  ->  two contiguous 8-element (16B) loads.
__device__ __forceinline__ v16bf afrag(const __bf16* rowp, int hi) {
  v8bf lo = *(const v8bf*)(rowp + hi * 16);
  v8bf hh = *(const v8bf*)(rowp + hi * 16 + 8);
  v16bf a;
#pragma unroll
  for (int i = 0; i < 8; ++i) { a[i] = lo[i]; a[i + 8] = hh[i]; }
  return a;
}

// B-fragment (32x16, 16-bit): N = lane&15, K = e + 16*hi -> 16 contiguous bf16.
__device__ __forceinline__ v16bf bfrag(const __bf16* p) {
  v8bf lo = *(const v8bf*)p;
  v8bf hh = *(const v8bf*)(p + 8);
  v16bf b;
#pragma unroll
  for (int i = 0; i < 8; ++i) { b[i] = lo[i]; b[i + 8] = hh[i]; }
  return b;
}

// Same B-fragment but sourced from fp32 memory (weight matrices), cvt to bf16.
__device__ __forceinline__ v16bf bfrag_f32(const float* p) {
  v16bf b;
  const float4* q = (const float4*)p;
#pragma unroll
  for (int j = 0; j < 4; ++j) {
    float4 t = q[j];
    b[4 * j + 0] = (__bf16)t.x; b[4 * j + 1] = (__bf16)t.y;
    b[4 * j + 2] = (__bf16)t.z; b[4 * j + 3] = (__bf16)t.w;
  }
  return b;
}

// Async DMA of one 16B segment from global memory into LDS (ASYNCcnt-tracked).
__device__ __forceinline__ void async_ld_b128(unsigned lds_byte_off, const void* g) {
  asm volatile("global_load_async_to_lds_b128 %0, %1, off"
               :
               : "v"(lds_byte_off), "v"((unsigned long long)(size_t)g)
               : "memory");
}
__device__ __forceinline__ void wait_async0() {
  asm volatile("s_wait_asynccnt 0x0" ::: "memory");
}

// Row reductions across the 16 lanes holding one C-fragment row (xor 1..8
// never crosses the lane<16 / lane>=16 halves, so rows stay separate).
__device__ __forceinline__ float redmax16(float v) {
  v = fmaxf(v, __shfl_xor(v, 1));
  v = fmaxf(v, __shfl_xor(v, 2));
  v = fmaxf(v, __shfl_xor(v, 4));
  v = fmaxf(v, __shfl_xor(v, 8));
  return v;
}
__device__ __forceinline__ float redsum16(float v) {
  v += __shfl_xor(v, 1);
  v += __shfl_xor(v, 2);
  v += __shfl_xor(v, 4);
  v += __shfl_xor(v, 8);
  return v;
}

// ---------------------------------------------------------------------------
// Kernel 1: q = (x Wq^T)*hd^-0.5, k = x Wk^T, v = x Wv^T   (bf16 outputs)
// q,k stored [b][head][t][hd]; v stored transposed [b][head][hd][t].
// ---------------------------------------------------------------------------
__global__ __launch_bounds__(256) void qkv_proj_kernel(
    const float* __restrict__ x, const float* __restrict__ Wq,
    const float* __restrict__ Wk, const float* __restrict__ Wv,
    __bf16* __restrict__ qws, __bf16* __restrict__ kws,
    __bf16* __restrict__ vtws) {
  __shared__ __bf16 xlds[16 * Hh];  // 16 KB of the WGP's 320 KB LDS

  const int tid = threadIdx.x;
  const int wave = tid >> 5, lane = tid & 31, hi = lane >> 4, ln = lane & 15;
  const int m0 = blockIdx.x * 16;  // row strip within flattened [B*T, H]

  const float4* xs = (const float4*)(x + (size_t)m0 * Hh);
  for (int i = tid; i < 16 * Hh / 4; i += 256) {
    float4 v = xs[i];
    int o = i * 4;
    xlds[o + 0] = (__bf16)v.x; xlds[o + 1] = (__bf16)v.y;
    xlds[o + 2] = (__bf16)v.z; xlds[o + 3] = (__bf16)v.w;
  }
  __syncthreads();

  for (int tile = wave; tile < 96; tile += 8) {
    const int mat = tile >> 5;          // 0=q 1=k 2=v
    const int n0 = (tile & 31) * 16;    // output-column tile
    const float* W = (mat == 0) ? Wq : ((mat == 1) ? Wk : Wv);
    // B(K,N) = W[n][k]; lane reads 16 contiguous fp32 along k.
    const float* wrow = W + (size_t)(n0 + ln) * Hh + hi * 16;

    v8f acc = {};
#pragma unroll 4
    for (int k0 = 0; k0 < Hh; k0 += 32) {
      v16bf a = afrag(&xlds[(size_t)ln * Hh + k0], hi);  // ds_load_b128 x2
      v16bf b = bfrag_f32(wrow + k0);                    // global_load_b128 x4
      acc = wmma_bf16(a, b, acc);
    }

    const float scl = (mat == 0) ? 0.17677669529663687f : 1.0f;  // hd^-0.5
    const int n = n0 + ln, head = n >> 5, hd = n & 31;
#pragma unroll
    for (int r = 0; r < 8; ++r) {
      int t = m0 + r + hi * 8;                 // C layout: row = r + 8*hi
      int bidx = t >> 11, tt = t & (Tt - 1);
      __bf16 val = (__bf16)(acc[r] * scl);
      if (mat == 0)
        qws[(((size_t)(bidx * NH + head)) * Tt + tt) * HD + hd] = val;
      else if (mat == 1)
        kws[(((size_t)(bidx * NH + head)) * Tt + tt) * HD + hd] = val;
      else
        vtws[(((size_t)(bidx * NH + head)) * HD + hd) * Tt + tt] = val;
    }
  }
}

// ---------------------------------------------------------------------------
// Kernel 2: flash-style attention. One block = one (b,h) x 128 query rows;
// each of 8 waves owns a 16-row Q tile; the whole block shares each 32-key
// chunk, which is staged once into LDS by global_load_async_to_lds_b128
// (double-buffered, ASYNCcnt) instead of 8x-redundant per-wave global loads.
// Bias (the 512 MiB stream) goes straight into the WMMA C operand with both
// masks fused; softmax is online; P re-layouts C->A via wave-private LDS.
// ---------------------------------------------------------------------------
__global__ __launch_bounds__(256) void attn_kernel(
    const __bf16* __restrict__ qws, const __bf16* __restrict__ kws,
    const __bf16* __restrict__ vtws, const float* __restrict__ bias,
    const unsigned char* __restrict__ amask,
    const unsigned char* __restrict__ pmask, float* __restrict__ out) {
  __shared__ __bf16 plds[8][16 * 32];   // one 16x32 P tile per wave (8 KB)
  // Double-buffered K/V chunk stage. Per buffer (2048 bf16 = 4 KB):
  //   [0    .. 1023] : K chunk,  [key 0..31][hd 0..31]
  //   [1024 .. 2047] : V^T chunk,[hd 0..31][key 0..31]
  __shared__ __bf16 stage[2][2048];

  const int tid = threadIdx.x;
  const int wave = tid >> 5, lane = tid & 31, hi = lane >> 4, ln = lane & 15;
  const int bh = blockIdx.x >> 4;      // 0..31 = b*16+h
  const int qblk = blockIdx.x & 15;
  const int b = bh >> 4, h = bh & 15;
  const int q0 = qblk * 128 + wave * 16;

  const __bf16* qp = qws + ((size_t)bh * Tt + q0) * HD;
  const __bf16* kp = kws + (size_t)bh * Tt * HD;
  const __bf16* vp = vtws + (size_t)bh * HD * Tt;
  const float* bp = bias + (size_t)bh * Tt * Tt;
  const unsigned char* pm = pmask + (size_t)b * Tt;

  // This thread's 16B async segment: tid<128 -> K chunk, else V^T chunk.
  const bool segK = tid < 128;
  const int sub = segK ? tid : tid - 128;
  const unsigned stage_base = (unsigned)(size_t)&stage[0][0];  // LDS byte addr
  const unsigned seg_loff = (segK ? 0u : 2048u) + (unsigned)sub * 16u;

  auto issue_chunk = [&](int s, int buf) {
    const char* g;
    if (segK)  // K chunk is 2 KB contiguous in global: rows of 32 bf16, keys consecutive
      g = (const char*)(kp + (size_t)s * HD) + (size_t)sub * 16;
    else       // V^T chunk: 32 rows (hd), 64B each, row stride Tt*2 bytes
      g = (const char*)(vp + s) + (size_t)(sub >> 2) * (Tt * 2) + (size_t)(sub & 3) * 16;
    async_ld_b128(stage_base + (unsigned)buf * 4096u + seg_loff, g);
  };

  v16bf qf = afrag(qp + (size_t)ln * HD, hi);  // resident Q fragment
  v8f o0 = {}, o1 = {};                        // O accum, hd 0..15 / 16..31
  float mrun[8], lrun[8];
#pragma unroll
  for (int r = 0; r < 8; ++r) { mrun[r] = -3.0e38f; lrun[r] = 0.0f; }

  __bf16* pl = &plds[wave][0];

  issue_chunk(0, 0);  // prologue: DMA chunk 0 into buffer 0

  for (int i = 0; i < Tt / 32; ++i) {
    const int s = i * 32;
    const __bf16* kc = &stage[i & 1][0];
    const __bf16* vc = &stage[i & 1][1024];

    // Bias tile -> C operand with masks fused (independent of the async DMA,
    // so these long-latency global loads overlap the LDS fill).
    const int c0 = s + ln, c1 = s + 16 + ln;
    const unsigned int pm0 = pm[c0], pm1 = pm[c1];
    v8f cb0, cb1;
#pragma unroll
    for (int r = 0; r < 8; ++r) {
      const size_t qr = (size_t)(q0 + r + hi * 8);
      const float* brow = bp + qr * Tt;
      const unsigned char* arow = amask + qr * Tt;
      cb0[r] = (arow[c0] | pm0) ? -__builtin_inff() : brow[c0];
      cb1[r] = (arow[c1] | pm1) ? -__builtin_inff() : brow[c1];
    }
    if (s + 32 < Tt)  // global_prefetch_b8 of next bias chunk
      __builtin_prefetch(bp + (size_t)(q0 + hi * 8) * Tt + s + 32 + ln, 0, 0);

    wait_async0();    // this wave's chunk-i segments have landed in LDS
    __syncthreads();  // publish chunk i; also proves chunk i-1 fully consumed
    if (i + 1 < Tt / 32) issue_chunk(i + 1, (i + 1) & 1);  // overlap with compute

    // S = Q K^T + bias : one WMMA per 16x16 score tile (hd==32==K).
    v16bf kf0 = bfrag(kc + (size_t)ln * HD + hi * 16);
    v16bf kf1 = bfrag(kc + (size_t)(16 + ln) * HD + hi * 16);
    v8f s0 = wmma_bf16(qf, kf0, cb0);
    v8f s1 = wmma_bf16(qf, kf1, cb1);

    // Online softmax per row; write P (bf16) to wave-private LDS tile.
#pragma unroll
    for (int r = 0; r < 8; ++r) {
      float a0 = s0[r], a1 = s1[r];
      float mx = redmax16(fmaxf(a0, a1));
      float mn = fmaxf(mrun[r], mx);
      float corr = __expf(mrun[r] - mn);
      float e0 = __expf(a0 - mn), e1 = __expf(a1 - mn);
      float rs = redsum16(e0 + e1);
      lrun[r] = lrun[r] * corr + rs;
      mrun[r] = mn;
      o0[r] *= corr; o1[r] *= corr;
      const int row = r + hi * 8;
      pl[row * 32 + ln] = (__bf16)e0;
      pl[row * 32 + 16 + ln] = (__bf16)e1;
    }
    asm volatile("s_wait_dscnt 0x0" ::: "memory");  // LDS store->load (same wave)

    // O += P V : P as A-fragment from LDS, V^T rows give contiguous B-frags.
    v16bf pf = afrag(pl + (size_t)ln * 32, hi);
    v16bf vf0 = bfrag(vc + (size_t)ln * 32 + hi * 16);
    v16bf vf1 = bfrag(vc + (size_t)(16 + ln) * 32 + hi * 16);
    o0 = wmma_bf16(pf, vf0, o0);
    o1 = wmma_bf16(pf, vf1, o1);
  }

#pragma unroll
  for (int r = 0; r < 8; ++r) {
    const size_t qr = (size_t)(q0 + r + hi * 8);
    float inv = 1.0f / lrun[r];
    float* orow = out + ((size_t)b * Tt + qr) * Hh + h * HD;
    orow[ln] = o0[r] * inv;
    orow[16 + ln] = o1[r] * inv;
  }
}

// ---------------------------------------------------------------------------
extern "C" void kernel_launch(void* const* d_in, const int* in_sizes, int n_in,
                              void* d_out, int out_size, void* d_ws,
                              size_t ws_size, hipStream_t stream) {
  (void)in_sizes; (void)n_in; (void)out_size; (void)ws_size;
  const float* x = (const float*)d_in[0];
  const float* bias = (const float*)d_in[1];
  const unsigned char* amask = (const unsigned char*)d_in[2];  // bool [T,T]
  const unsigned char* pmask = (const unsigned char*)d_in[3];  // bool [B,T]
  const float* Wq = (const float*)d_in[4];
  const float* Wk = (const float*)d_in[5];
  const float* Wv = (const float*)d_in[6];
  float* out = (float*)d_out;

  __bf16* qws = (__bf16*)d_ws;          // 4 MiB each: q, k, v^T in bf16
  __bf16* kws = qws + PER_MAT;
  __bf16* vtws = kws + PER_MAT;

  qkv_proj_kernel<<<dim3(Bb * Tt / 16), dim3(256), 0, stream>>>(
      x, Wq, Wk, Wv, qws, kws, vtws);
  attn_kernel<<<dim3(Bb * NH * (Tt / 128)), dim3(256), 0, stream>>>(
      qws, kws, vtws, bias, amask, pmask, out);
}
